// BertSelfAttention_14053132993044
// MI455X (gfx1250) — matrix operand order repaired and checked by reading, not compiled, not run on hardware
//
#include <hip/hip_runtime.h>
#include <hip/hip_bf16.h>

#define B_  2
#define S_  2048
#define D_  1024
#define H_  16
#define HD_ 64

typedef _Float16 half8  __attribute__((ext_vector_type(8)));
typedef _Float16 half16 __attribute__((ext_vector_type(16)));
typedef float    f32x8  __attribute__((ext_vector_type(8)));

static __device__ __forceinline__ half16 combine8(half8 lo, half8 hi) {
  return __builtin_shufflevector(lo, hi, 0,1,2,3,4,5,6,7,8,9,10,11,12,13,14,15);
}

// CDNA5 async global->LDS copy (ASYNCcnt). GVS mode: saddr(SGPR64) + voffset(VGPR32),
// LDS byte offset in a VGPR.
static __device__ __forceinline__ void async_b128(unsigned int lds_byte_off,
                                                  unsigned long long saddr,
                                                  unsigned int voff_bytes) {
  asm volatile("global_load_async_to_lds_b128 %0, %1, %2"
               :
               : "v"(lds_byte_off), "v"(voff_bytes), "s"(saddr)
               : "memory");
}
#define WAIT_ASYNC(n) asm volatile("s_wait_asynccnt " #n ::: "memory")

static __device__ __forceinline__ unsigned int lds_off_u32(const void* p) {
  // flat LDS address: [63:32]=aperture, [31:0]=wave-relative LDS offset
  return (unsigned int)(uintptr_t)p;
}

// ---------------------------------------------------------------------------
// fp32 -> fp16 conversion (X and the three weight matrices)
// ---------------------------------------------------------------------------
__global__ void cvt_f32_f16(const float* __restrict__ src,
                            _Float16* __restrict__ dst, int n) {
  int i = blockIdx.x * blockDim.x + threadIdx.x;
  if (i < n) dst[i] = (_Float16)src[i];
}

// ---------------------------------------------------------------------------
// QKV projection: out = X @ W^T + b, written head-split in f16.
//   z = 0: Q  -> [B,H,S,HD]  (scaled by 1/sqrt(HD))
//   z = 1: K  -> [B,H,S,HD]
//   z = 2: V  -> [B,H,HD,S]  (transposed so PV WMMA B-frag is K-contiguous)
// Block: 256 threads = 8 waves, tile 256(M) x 64(N); wave w owns 32 rows.
// W tile (64x32 f16 = 4KB) staged in LDS via async copies, double-buffered.
// ---------------------------------------------------------------------------
__global__ __launch_bounds__(256)
void qkv_wmma(const _Float16* __restrict__ Xh,   // [B*S, D]
              const _Float16* __restrict__ Wh,   // [3][D, D] (row n along k)
              const float* __restrict__ bq,
              const float* __restrict__ bk,
              const float* __restrict__ bv,
              _Float16* __restrict__ Qh,
              _Float16* __restrict__ Kh,
              _Float16* __restrict__ Vt) {
  __shared__ __align__(32) _Float16 ldsW[2][64 * 32];   // 2 x 4KB

  const int z    = blockIdx.z;
  const int m0   = blockIdx.x * 256;
  const int n0   = blockIdx.y * 64;
  const int tid  = threadIdx.x;
  const int wave = tid >> 5;
  const int lane = tid & 31;
  const int l16  = lane & 15;
  const int grp  = lane >> 4;

  const _Float16* W    = Wh + (size_t)z * D_ * D_;
  const float*    bias = (z == 0) ? bq : (z == 1) ? bk : bv;

  // async-copy assignment: 256 threads x 16B cover the 64x32 W tile
  const int wn  = tid >> 2;                        // tile row (n_local) 0..63
  const int wsg = tid & 3;                         // 16B segment 0..3
  const unsigned long long wsad = (unsigned long long)(uintptr_t)W;
  const unsigned int wvoff = (unsigned int)(((n0 + wn) * D_ + wsg * 8) * 2);
  const unsigned int wldso = (unsigned int)(wn * 64 + wsg * 16);
  const unsigned int ldsb[2] = { lds_off_u32(&ldsW[0][0]), lds_off_u32(&ldsW[1][0]) };

  const _Float16* aptr0 = Xh + (size_t)(m0 + wave * 32 + l16) * D_;
  const _Float16* aptr1 = aptr0 + (size_t)16 * D_;

  f32x8 acc[2][4];
#pragma unroll
  for (int s = 0; s < 2; ++s)
#pragma unroll
    for (int j = 0; j < 4; ++j) acc[s][j] = (f32x8){};

  // prologue: stage K-step 0 into buffer 0
  async_b128(ldsb[0] + wldso, wsad, wvoff);

  const int NK = D_ / 32;
  for (int t = 0; t < NK; ++t) {
    const int kb = t * 32;
    if (t + 1 < NK) {
      async_b128(ldsb[(t + 1) & 1] + wldso, wsad,
                 wvoff + (unsigned int)((kb + 32) * 2));
      WAIT_ASYNC(1);          // current buffer complete (in-order)
    } else {
      WAIT_ASYNC(0);
    }
    __syncthreads();

    // A fragments (16x32 f16) for both M sub-tiles, from global
    half16 af[2];
#pragma unroll
    for (int s = 0; s < 2; ++s) {
      const _Float16* ap = (s == 0) ? aptr0 : aptr1;
      half8 lo = *(const half8*)(ap + kb + grp * 8);
      half8 hi = *(const half8*)(ap + kb + 16 + grp * 8);
      af[s] = combine8(lo, hi);
    }
    // B fragments from LDS, shared by all 8 waves
    const _Float16* wl = &ldsW[t & 1][0];
#pragma unroll
    for (int j = 0; j < 4; ++j) {
      half16 bf = combine8(*(const half8*)(wl + (j * 16 + l16) * 32 + grp * 8),
                           *(const half8*)(wl + (j * 16 + l16) * 32 + 16 + grp * 8));
#pragma unroll
      for (int s = 0; s < 2; ++s)
        acc[s][j] = __builtin_amdgcn_wmma_f32_16x16x32_f16(
            false, af[s], false, bf, (short)0, acc[s][j], false, false);
    }
    __syncthreads();          // protect buffer reuse against next issue
  }

  // Epilogue: C layout -> row = v + 8*grp, col = n0 + j*16 + l16
#pragma unroll
  for (int j = 0; j < 4; ++j) {
    const int n  = n0 + j * 16 + l16;
    const float bb = bias[n];
    const int h  = n >> 6;
    const int hd = n & 63;
#pragma unroll
    for (int s = 0; s < 2; ++s) {
#pragma unroll
      for (int v = 0; v < 8; ++v) {
        const int m = m0 + wave * 32 + s * 16 + v + 8 * grp;
        const int b = m >> 11;
        const int sq = m & (S_ - 1);
        const float val = acc[s][j][v] + bb;
        if (z == 0) {
          Qh[((size_t)(b * H_ + h) * S_ + sq) * HD_ + hd] = (_Float16)(val * 0.125f);
        } else if (z == 1) {
          Kh[((size_t)(b * H_ + h) * S_ + sq) * HD_ + hd] = (_Float16)val;
        } else {
          Vt[((size_t)(b * H_ + h) * HD_ + hd) * S_ + sq] = (_Float16)val;
        }
      }
    }
  }
}

// ---------------------------------------------------------------------------
// Flash-style attention: one block = (q-tile of 64 rows, h, b); 4 waves,
// each wave owns 16 q rows, streams keys in 64-wide chunks.
// K chunk (64x64) and V chunk (64x64, transposed layout) are staged in LDS
// via double-buffered async copies shared by all 4 waves.
// ---------------------------------------------------------------------------
__global__ __launch_bounds__(128)
void attn_wmma(const _Float16* __restrict__ Qh,   // [B,H,S,HD]
               const _Float16* __restrict__ Kh,   // [B,H,S,HD]
               const _Float16* __restrict__ Vt,   // [B,H,HD,S]
               const float* __restrict__ mask,    // [B,1,1,S]
               float* __restrict__ out) {         // [B,S,D]
  __shared__ __align__(32) _Float16 ldsK[2][64 * 64];   // 2 x 8KB
  __shared__ __align__(32) _Float16 ldsV[2][64 * 64];   // 2 x 8KB
  __shared__ __align__(32) _Float16 ldsP[4 * 16 * 64];  // 8KB (P staging)

  const int b    = blockIdx.z;
  const int h    = blockIdx.y;
  const int bh   = b * H_ + h;
  const int tid  = threadIdx.x;
  const int wave = tid >> 5;
  const int lane = tid & 31;
  const int l16  = lane & 15;
  const int grp  = lane >> 4;
  const int q0   = blockIdx.x * 64 + wave * 16;

  const unsigned long long ksad =
      (unsigned long long)(uintptr_t)(Kh + (size_t)bh * S_ * HD_);
  const unsigned long long vsad =
      (unsigned long long)(uintptr_t)(Vt + (size_t)bh * HD_ * S_);
  const unsigned int kbuf[2] = { lds_off_u32(&ldsK[0][0]), lds_off_u32(&ldsK[1][0]) };
  const unsigned int vbuf[2] = { lds_off_u32(&ldsV[0][0]), lds_off_u32(&ldsV[1][0]) };

  // stage one 64-key chunk (K rows: key-major, V rows: hd-major): 8 asyncs/thread
  auto issue_chunk = [&](int kb, int bsel) {
#pragma unroll
    for (int i = 0; i < 4; ++i) {
      const int idx = tid + 128 * i;          // 0..511 segment id
      const int row = idx >> 3;               // 0..63
      const int seg = idx & 7;                // 16B segment within 128B row
      async_b128(kbuf[bsel] + (unsigned int)(row * 128 + seg * 16), ksad,
                 (unsigned int)(((kb + row) * HD_ + seg * 8) * 2));
      async_b128(vbuf[bsel] + (unsigned int)(row * 128 + seg * 16), vsad,
                 (unsigned int)((row * S_ + kb + seg * 8) * 2));
    }
  };

  // Q A-fragments for this wave's 16 rows (HD=64 -> two K=32 fragments)
  const _Float16* qrow = Qh + ((size_t)bh * S_ + q0 + l16) * HD_;
  half16 qf[2];
#pragma unroll
  for (int kk = 0; kk < 2; ++kk) {
    half8 lo = *(const half8*)(qrow + kk * 32 + grp * 8);
    half8 hi = *(const half8*)(qrow + kk * 32 + 16 + grp * 8);
    qf[kk] = combine8(lo, hi);
  }

  float mrow[8], lrow[8];
  f32x8 o[4];
#pragma unroll
  for (int v = 0; v < 8; ++v) { mrow[v] = -3.0e38f; lrow[v] = 0.0f; }
#pragma unroll
  for (int n = 0; n < 4; ++n) o[n] = (f32x8){};

  _Float16* pl = ldsP + wave * 16 * 64;            // wave-private P staging
  const float* bmask = mask + (size_t)b * S_;

  issue_chunk(0, 0);

  const int NC = S_ / 64;
  for (int c = 0; c < NC; ++c) {
    const int kb = c * 64;
    if (c + 1 < NC) {
      issue_chunk(kb + 64, (c + 1) & 1);
      WAIT_ASYNC(8);          // current chunk's 8 copies complete (in-order)
    } else {
      WAIT_ASYNC(0);
    }
    __syncthreads();

    const _Float16* kl = &ldsK[c & 1][0];
    const _Float16* vl = &ldsV[c & 1][0];

    // ---- scores S[16 x 64] = Q Kt (+mask), C-layout accumulators ----
    f32x8 s[4];
#pragma unroll
    for (int j = 0; j < 4; ++j) {
      f32x8 a = (f32x8){};
#pragma unroll
      for (int kk = 0; kk < 2; ++kk) {
        half16 bf = combine8(*(const half8*)(kl + (j * 16 + l16) * 64 + kk * 32 + grp * 8),
                             *(const half8*)(kl + (j * 16 + l16) * 64 + kk * 32 + 16 + grp * 8));
        a = __builtin_amdgcn_wmma_f32_16x16x32_f16(
            false, qf[kk], false, bf, (short)0, a, false, false);
      }
      s[j] = a + bmask[kb + j * 16 + l16];
    }

    // ---- online softmax per row (row = v + 8*grp, 16-lane reductions) ----
    float al[8];
#pragma unroll
    for (int v = 0; v < 8; ++v) {
      float cm = fmaxf(fmaxf(s[0][v], s[1][v]), fmaxf(s[2][v], s[3][v]));
#pragma unroll
      for (int off = 1; off < 16; off <<= 1) cm = fmaxf(cm, __shfl_xor(cm, off));
      const float mnew = fmaxf(mrow[v], cm);
      al[v]   = __expf(mrow[v] - mnew);
      mrow[v] = mnew;
      float rs = 0.0f;
#pragma unroll
      for (int j = 0; j < 4; ++j) {
        const float p = __expf(s[j][v] - mnew);
        s[j][v] = p;
        rs += p;
      }
#pragma unroll
      for (int off = 1; off < 16; off <<= 1) rs += __shfl_xor(rs, off);
      lrow[v] = lrow[v] * al[v] + rs;
    }
#pragma unroll
    for (int n = 0; n < 4; ++n)
#pragma unroll
      for (int v = 0; v < 8; ++v) o[n][v] *= al[v];

    // ---- P: C-layout -> A-layout via wave-private LDS (in-order per wave) ----
#pragma unroll
    for (int j = 0; j < 4; ++j)
#pragma unroll
      for (int v = 0; v < 8; ++v)
        pl[(v + 8 * grp) * 64 + j * 16 + l16] = (_Float16)s[j][v];

    half16 pa[2];
#pragma unroll
    for (int kk = 0; kk < 2; ++kk) {
      half8 lo = *(const half8*)(pl + l16 * 64 + kk * 32 + grp * 8);
      half8 hi = *(const half8*)(pl + l16 * 64 + kk * 32 + 16 + grp * 8);
      pa[kk] = combine8(lo, hi);
    }

    // ---- O += P @ V  (ldsV rows are hd, K(=key) contiguous) ----
#pragma unroll
    for (int n = 0; n < 4; ++n) {
#pragma unroll
      for (int kk = 0; kk < 2; ++kk) {
        half16 bf = combine8(*(const half8*)(vl + (n * 16 + l16) * 64 + kk * 32 + grp * 8),
                             *(const half8*)(vl + (n * 16 + l16) * 64 + kk * 32 + 16 + grp * 8));
        o[n] = __builtin_amdgcn_wmma_f32_16x16x32_f16(
            false, pa[kk], false, bf, (short)0, o[n], false, false);
      }
    }
    __syncthreads();          // protect K/V buffer reuse against next issue
  }

  // ---- epilogue: O / l, merge heads -> [B,S,D] fp32 ----
#pragma unroll
  for (int n = 0; n < 4; ++n) {
    const int d = h * HD_ + n * 16 + l16;
#pragma unroll
    for (int v = 0; v < 8; ++v) {
      const int row = q0 + v + 8 * grp;
      out[((size_t)b * S_ + row) * D_ + d] = o[n][v] / lrow[v];
    }
  }
}

// ---------------------------------------------------------------------------
extern "C" void kernel_launch(void* const* d_in, const int* in_sizes, int n_in,
                              void* d_out, int out_size, void* d_ws, size_t ws_size,
                              hipStream_t stream) {
  const float* hs   = (const float*)d_in[0];
  const float* mask = (const float*)d_in[1];
  const float* Wq   = (const float*)d_in[2];
  const float* bq   = (const float*)d_in[3];
  const float* Wk   = (const float*)d_in[4];
  const float* bk   = (const float*)d_in[5];
  const float* Wv   = (const float*)d_in[6];
  const float* bv   = (const float*)d_in[7];
  float* out = (float*)d_out;

  // Workspace layout (38 MB total)
  char* ws = (char*)d_ws;
  _Float16* Xh = (_Float16*)(ws);                          //  8 MB  [B*S, D]
  _Float16* Wh = (_Float16*)(ws + (size_t)8  * (1u<<20));  //  6 MB  [3][D,D]
  _Float16* Qh = (_Float16*)(ws + (size_t)14 * (1u<<20));  //  8 MB
  _Float16* Kh = (_Float16*)(ws + (size_t)22 * (1u<<20));  //  8 MB
  _Float16* Vt = (_Float16*)(ws + (size_t)30 * (1u<<20));  //  8 MB

  const int nX = B_ * S_ * D_;   // 4,194,304
  const int nW = D_ * D_;        // 1,048,576

  cvt_f32_f16<<<nX / 256, 256, 0, stream>>>(hs, Xh, nX);
  cvt_f32_f16<<<nW / 256, 256, 0, stream>>>(Wq, Wh,          nW);
  cvt_f32_f16<<<nW / 256, 256, 0, stream>>>(Wk, Wh + nW,     nW);
  cvt_f32_f16<<<nW / 256, 256, 0, stream>>>(Wv, Wh + 2 * nW, nW);

  qkv_wmma<<<dim3((B_ * S_) / 256, D_ / 64, 3), 256, 0, stream>>>(
      Xh, Wh, bq, bk, bv, Qh, Kh, Vt);

  attn_wmma<<<dim3(S_ / 64, H_, B_), 128, 0, stream>>>(
      Qh, Kh, Vt, mask, out);
}
